// MultiHeadAttention_21062519620032
// MI455X (gfx1250) — compile-verified, matrix operand order checked by reading, not run on hardware
//
#include <hip/hip_runtime.h>

// ---------------------------------------------------------------------------
// MI455X (gfx1250) multi-head attention, bf16 WMMA + flash-attention fusion,
// async-to-LDS double-buffered GEMMs.
// b=2, L=2048, d_model=1024, n_head=16, d_head=64.
// ---------------------------------------------------------------------------

typedef __bf16 bf16_t;
typedef __attribute__((ext_vector_type(8)))  bf16_t v8bf;
typedef __attribute__((ext_vector_type(16))) bf16_t v16bf;
typedef __attribute__((ext_vector_type(8)))  float  v8f;

#define WMMA_BF16(A, B, C) \
  __builtin_amdgcn_wmma_f32_16x16x32_bf16(false, (A), false, (B), (short)0, (C), false, false)

// Load one 16x32 bf16 A/B^T fragment (ISA 16-bit A-matrix layout, wave32):
// lane (0-15): row m = lane, halfwords = K[0..7], K[16..23]
// lane (16-31): row m = lane-16, halfwords = K[8..15], K[24..31]
// p points at (row m, k-base 0) of a K-contiguous row; hi = lane>>4.
// Two 16-byte loads per lane -> global_load_b128 / ds_load_b128 x2.
static __device__ __forceinline__ v16bf load_frag(const bf16_t* p, int hi) {
  union { v16bf f; struct { v8bf lo; v8bf up; } h; } u;
  u.h.lo = *reinterpret_cast<const v8bf*>(p + hi * 8);
  u.h.up = *reinterpret_cast<const v8bf*>(p + 16 + hi * 8);
  return u.f;
}

// gfx1250 async copy global->LDS, 16 bytes per lane, tracked by ASYNCcnt.
// VGLOBAL encoding: vdst = LDS byte offset VGPR, vaddr = 64-bit address.
static __device__ __forceinline__ void async_load_b128(unsigned int lds_off,
                                                       const bf16_t* g) {
  asm volatile("global_load_async_to_lds_b128 %0, %1, off"
               :: "v"(lds_off), "v"((unsigned long long)(uintptr_t)g)
               : "memory");
}

// xor-shuffle within 16-lane row groups via ds_swizzle (group-of-32 encoding:
// offset = {1'b0, xor[14:10], or[9:5], and[4:0]}). Masks 1/2/4/8 stay in-half.
template <int XMASK>
static __device__ __forceinline__ float swz_xor(float v) {
  int i = __builtin_bit_cast(int, v);
  i = __builtin_amdgcn_ds_swizzle(i, (XMASK << 10) | 0x1f);
  return __builtin_bit_cast(float, i);
}

// ---------------------------------------------------------------------------
// Conversion / layout kernels
// ---------------------------------------------------------------------------
__global__ void convert_f32_bf16(const float* __restrict__ in, bf16_t* __restrict__ out, int n) {
  int i = blockIdx.x * blockDim.x + threadIdx.x;
  if (i < n) out[i] = (bf16_t)in[i];
}

// w_{q,k,v}: [n,m,d] f32 -> Wt[n*64+d][m] bf16 (i.e. B^T with K=m contiguous)
__global__ void permute_qkv(const float* __restrict__ wq, const float* __restrict__ wk,
                            const float* __restrict__ wv, bf16_t* __restrict__ oq,
                            bf16_t* __restrict__ ok, bf16_t* __restrict__ ov) {
  int i = blockIdx.x * blockDim.x + threadIdx.x;  // 16*1024*64 = n*65536 + m*64 + d
  if (i >= 16 * 1024 * 64) return;
  int d  = i & 63;
  int m  = (i >> 6) & 1023;
  int nh = i >> 16;
  size_t dst = (size_t)(nh * 64 + d) * 1024 + m;
  oq[dst] = (bf16_t)wq[i];
  ok[dst] = (bf16_t)wk[i];
  ov[dst] = (bf16_t)wv[i];
}

// w_o: [n,d,m] f32 (flat 1024x1024, rows n*64+d) -> WOt[m][n*64+d] bf16 (B^T)
__global__ void transpose_wo(const float* __restrict__ in, bf16_t* __restrict__ out) {
  int i = blockIdx.x * blockDim.x + threadIdx.x;  // 1024*1024
  if (i >= 1024 * 1024) return;
  int c = i & 1023;  // m
  int r = i >> 10;   // n*64+d
  out[(size_t)c * 1024 + r] = (bf16_t)in[i];
}

// ---------------------------------------------------------------------------
// bf16 GEMM: C[M,N] = A[M,K] @ B[K,N], with B supplied transposed Bt[N][K].
// Block 256 threads = 8 waves, block tile 128x128, wave tile 64x32.
// K-step 32, double-buffered LDS tiles filled with global_load_async_to_lds,
// pipelined with s_wait_asynccnt (4 async issues per tile per wave, in-order
// completion => "<=4" means the current tile has landed while the next one
// is still in flight).
// OUT_MODE: 0 = bf16 row-major, 1 = bf16 transposed (for V^T), 2 = f32 row-major
// ---------------------------------------------------------------------------
template <int OUT_MODE>
__global__ __launch_bounds__(256) void gemm_bf16(const bf16_t* __restrict__ A,
                                                 const bf16_t* __restrict__ Bt,
                                                 void* __restrict__ Cout,
                                                 int M, int N, int K) {
  __shared__ bf16_t lA[2][128 * 32];  // 8 KB per buffer
  __shared__ bf16_t lB[2][128 * 32];  // 8 KB per buffer

  const int t    = threadIdx.x;
  const int lane = t & 31;
  const int wv   = t >> 5;
  const int wm   = wv & 1;   // 2 waves along M
  const int wn   = wv >> 1;  // 4 waves along N
  const int n    = lane & 15;
  const int hi   = lane >> 4;
  const int rowB = blockIdx.y * 128;
  const int colB = blockIdx.x * 128;
  const int row0 = rowB + wm * 64;
  const int col0 = colB + wn * 32;

  v8f acc[4][2] = {};

  // Tile fill: 128 rows x 32 k of A and Bt = 512 chunks of 16B each;
  // 256 threads -> 2 chunks each per matrix -> 4 async issues per thread.
  auto issue = [&](int buf, int k) {
#pragma unroll
    for (int i = 0; i < 2; ++i) {
      int c  = t + 256 * i;
      int r  = c >> 2;          // row within tile
      int c8 = (c & 3) * 8;     // k-offset within row
      async_load_b128((unsigned int)(uintptr_t)&lA[buf][c * 8],
                      A + (size_t)(rowB + r) * K + k + c8);
      async_load_b128((unsigned int)(uintptr_t)&lB[buf][c * 8],
                      Bt + (size_t)(colB + r) * K + k + c8);
    }
  };

  issue(0, 0);
  const int KT = K / 32;
  for (int kt = 0; kt < KT; ++kt) {
    const int buf = kt & 1;
    if (kt + 1 < KT) {
      issue(buf ^ 1, (kt + 1) * 32);
      asm volatile("s_wait_asynccnt 0x4" ::: "memory");  // current tile landed
    } else {
      asm volatile("s_wait_asynccnt 0x0" ::: "memory");
    }
    __syncthreads();

    v16bf af[4], bf[2];
#pragma unroll
    for (int mi = 0; mi < 4; ++mi)
      af[mi] = load_frag(&lA[buf][(wm * 64 + mi * 16 + n) * 32], hi);
#pragma unroll
    for (int ni = 0; ni < 2; ++ni)
      bf[ni] = load_frag(&lB[buf][(wn * 32 + ni * 16 + n) * 32], hi);
#pragma unroll
    for (int mi = 0; mi < 4; ++mi)
#pragma unroll
      for (int ni = 0; ni < 2; ++ni)
        acc[mi][ni] = WMMA_BF16(af[mi], bf[ni], acc[mi][ni]);

    __syncthreads();  // safe to overwrite this buffer next round
  }

  // C layout: lane l, vgpr v -> row = v + 8*(l>=16), col = l&15 (per 16x16 tile)
#pragma unroll
  for (int mi = 0; mi < 4; ++mi)
#pragma unroll
    for (int ni = 0; ni < 2; ++ni)
#pragma unroll
      for (int v = 0; v < 8; ++v) {
        int r = row0 + mi * 16 + v + 8 * hi;
        int c = col0 + ni * 16 + n;
        float val = acc[mi][ni][v];
        if (OUT_MODE == 2)      ((float*)Cout)[(size_t)r * N + c] = val;
        else if (OUT_MODE == 1) ((bf16_t*)Cout)[(size_t)c * M + r] = (bf16_t)val;
        else                    ((bf16_t*)Cout)[(size_t)r * N + c] = (bf16_t)val;
      }
}

// ---------------------------------------------------------------------------
// Flash attention: one wave per 16-row Q tile of one (batch, head).
// Q,K row-major [4096][1024] bf16 (head-contiguous cols); V^T [1024][4096].
// Per 32-key step: 4 score WMMAs + online softmax + 4 ctx WMMAs.
// Scores/softmax/P never touch HBM (512 MB avoided vs naive).
// ---------------------------------------------------------------------------
__global__ __launch_bounds__(128) void flash_attn(const bf16_t* __restrict__ Qb,
                                                  const bf16_t* __restrict__ Kb,
                                                  const bf16_t* __restrict__ Vt,
                                                  const unsigned char* __restrict__ mask,
                                                  bf16_t* __restrict__ Cb) {
  __shared__ bf16_t plds[4][16][32];  // per-wave P staging (C-layout -> A-layout)

  const int lane  = threadIdx.x & 31;
  const int wv    = threadIdx.x >> 5;
  const int head  = blockIdx.y;
  const int batch = blockIdx.z;
  const int qrow0 = (blockIdx.x * 4 + wv) * 16;
  const int n     = lane & 15;
  const int hi    = lane >> 4;
  const size_t batchL = (size_t)batch * 2048;

  const bf16_t* qp = Qb + (batchL + qrow0 + n) * 1024 + head * 64;
  v16bf aq0 = load_frag(qp, hi);        // d 0..31
  v16bf aq1 = load_frag(qp + 32, hi);   // d 32..63

  float mrun[8], lrun[8];
  v8f o[4] = {};
#pragma unroll
  for (int v = 0; v < 8; ++v) { mrun[v] = -3.0e38f; lrun[v] = 0.0f; }

  for (int j = 0; j < 64; ++j) {
    const int key0 = j * 32;
    const bf16_t* kp = Kb + (batchL + key0 + n) * 1024 + head * 64;
    v16bf bk00 = load_frag(kp, hi);
    v16bf bk01 = load_frag(kp + 32, hi);
    v16bf bk10 = load_frag(kp + 16 * 1024, hi);
    v16bf bk11 = load_frag(kp + 16 * 1024 + 32, hi);

    v8f s0 = {}, s1 = {};
    s0 = WMMA_BF16(aq0, bk00, s0);
    s0 = WMMA_BF16(aq1, bk01, s0);
    s1 = WMMA_BF16(aq0, bk10, s1);
    s1 = WMMA_BF16(aq1, bk11, s1);

    const float msk0 = mask[batchL + key0 + n]      ? -1.0e9f : 0.0f;
    const float msk1 = mask[batchL + key0 + 16 + n] ? -1.0e9f : 0.0f;

    float p0[8], p1[8], cvec[8];
#pragma unroll
    for (int v = 0; v < 8; ++v) {
      float a = s0[v] * 0.125f + msk0;  // 1/sqrt(64) = 0.125
      float b = s1[v] * 0.125f + msk1;
      float rm = fmaxf(a, b);
      rm = fmaxf(rm, swz_xor<1>(rm));
      rm = fmaxf(rm, swz_xor<2>(rm));
      rm = fmaxf(rm, swz_xor<4>(rm));
      rm = fmaxf(rm, swz_xor<8>(rm));
      float mnew = fmaxf(mrun[v], rm);
      cvec[v] = __expf(mrun[v] - mnew);
      mrun[v] = mnew;
      p0[v] = __expf(a - mnew);
      p1[v] = __expf(b - mnew);
      float rs = p0[v] + p1[v];
      rs += swz_xor<1>(rs);
      rs += swz_xor<2>(rs);
      rs += swz_xor<4>(rs);
      rs += swz_xor<8>(rs);
      lrun[v] = lrun[v] * cvec[v] + rs;
    }
#pragma unroll
    for (int c = 0; c < 4; ++c)
#pragma unroll
      for (int v = 0; v < 8; ++v) o[c][v] *= cvec[v];

    // P: C-layout regs -> LDS row-major [16][32] -> A-layout frag.
    // Same-wave DS ops are in-order; compiler barrier prevents reordering.
#pragma unroll
    for (int v = 0; v < 8; ++v) {
      int row = v + 8 * hi;
      plds[wv][row][n]      = (bf16_t)p0[v];
      plds[wv][row][16 + n] = (bf16_t)p1[v];
    }
    asm volatile("" ::: "memory");
    v16bf pa;
    {
      union { v16bf f; struct { v8bf lo; v8bf up; } h; } u;
      u.h.lo = *reinterpret_cast<const v8bf*>(&plds[wv][n][hi * 8]);
      u.h.up = *reinterpret_cast<const v8bf*>(&plds[wv][n][16 + hi * 8]);
      pa = u.f;
    }
    asm volatile("" ::: "memory");

#pragma unroll
    for (int c = 0; c < 4; ++c) {
      const bf16_t* vp = Vt + (size_t)(head * 64 + c * 16 + n) * 4096 + batchL + key0;
      v16bf bvf = load_frag(vp, hi);
      o[c] = WMMA_BF16(pa, bvf, o[c]);
    }
  }

  // normalize and store ctx, head-contiguous layout [b*L][n*64+d]
#pragma unroll
  for (int v = 0; v < 8; ++v) {
    float inv = 1.0f / lrun[v];
    int row = v + 8 * hi;
    bf16_t* cp = Cb + (batchL + qrow0 + row) * 1024 + head * 64;
#pragma unroll
    for (int c = 0; c < 4; ++c) cp[c * 16 + n] = (bf16_t)(o[c][v] * inv);
  }
}

// ---------------------------------------------------------------------------
// Host-side orchestration
// ---------------------------------------------------------------------------
extern "C" void kernel_launch(void* const* d_in, const int* in_sizes, int n_in,
                              void* d_out, int out_size, void* d_ws, size_t ws_size,
                              hipStream_t stream) {
  const float* x  = (const float*)d_in[0];
  const float* wq = (const float*)d_in[1];
  const float* wk = (const float*)d_in[2];
  const float* wv = (const float*)d_in[3];
  const float* wo = (const float*)d_in[4];
  const unsigned char* mask = (const unsigned char*)d_in[5];
  (void)in_sizes; (void)n_in; (void)out_size; (void)ws_size;

  char* ws = (char*)d_ws;
  const size_t MB = 1024 * 1024;
  bf16_t* Xb  = (bf16_t*)(ws + 0);        // [4096][1024]
  bf16_t* Qb  = (bf16_t*)(ws + 8 * MB);   // [4096][1024]
  bf16_t* Kb  = (bf16_t*)(ws + 16 * MB);  // [4096][1024]
  bf16_t* Vt  = (bf16_t*)(ws + 24 * MB);  // [1024][4096]  (V transposed)
  bf16_t* Cb  = (bf16_t*)(ws + 32 * MB);  // [4096][1024]  (ctx)
  bf16_t* WQt = (bf16_t*)(ws + 40 * MB);  // [1024][1024]
  bf16_t* WKt = (bf16_t*)(ws + 42 * MB);
  bf16_t* WVt = (bf16_t*)(ws + 44 * MB);
  bf16_t* WOt = (bf16_t*)(ws + 46 * MB);

  convert_f32_bf16<<<(4194304 + 255) / 256, 256, 0, stream>>>(x, Xb, 4194304);
  permute_qkv<<<(1048576 + 255) / 256, 256, 0, stream>>>(wq, wk, wv, WQt, WKt, WVt);
  transpose_wo<<<(1048576 + 255) / 256, 256, 0, stream>>>(wo, WOt);

  dim3 g(1024 / 128, 4096 / 128);  // (8, 32), 256 threads/block
  gemm_bf16<0><<<g, 256, 0, stream>>>(Xb, WQt, Qb, 4096, 1024, 1024);
  gemm_bf16<0><<<g, 256, 0, stream>>>(Xb, WKt, Kb, 4096, 1024, 1024);
  gemm_bf16<1><<<g, 256, 0, stream>>>(Xb, WVt, Vt, 4096, 1024, 1024);

  flash_attn<<<dim3(32, 16, 2), 128, 0, stream>>>(Qb, Kb, Vt, mask, Cb);

  gemm_bf16<2><<<g, 256, 0, stream>>>(Cb, WOt, d_out, 4096, 1024, 1024);
}
